// GATClassifier_10625749090950
// MI455X (gfx1250) — compile-verified, matrix operand order checked by reading, not run on hardware
//
#include <hip/hip_runtime.h>
#include <hip/hip_bf16.h>

typedef __attribute__((ext_vector_type(16))) _Float16 v16h;
typedef __attribute__((ext_vector_type(8)))  float    v8f;

#define NNODES   50000
#define NEDGES   800000
#define NHEADS   4
#define NCH      32
#define NHC      128
#define NGRAPH   64
#define NCLS     4
#define NEG_SLOPE 0.2f

// j (0..15) -> k offset within a 32-wide K tile, for 16-bit A/B fragments.
// VGPR vg=j>>1 holds 2 halfs; vg 0..3 -> K = kgrp*8 + vg*2 + hi,
// vg 4..7 -> K = 16 + kgrp*8 + (vg-4)*2 + hi  (kgrp = lane>>4).
__device__ __forceinline__ int kmap(int kgrp, int j) {
  int vg = j >> 1, hi = j & 1;
  return (vg < 4) ? (kgrp * 8 + vg * 2 + hi)
                  : (16 + kgrp * 8 + (vg - 4) * 2 + hi);
}

// Accumulate one 16x16 output tile: 4 WMMAs along K=128.
__device__ __forceinline__ v8f tile_acc(const _Float16* sBs, const v16h* afrag,
                                        int lane, int nt) {
  v8f acc = {};
#pragma unroll
  for (int kt = 0; kt < 4; ++kt) {
    v16h bfrag = *(const v16h*)&sBs[((kt * 8 + nt) * 32 + lane) * 16];
    acc = __builtin_amdgcn_wmma_f32_16x16x32_f16(
        false, afrag[kt], false, bfrag, (short)0, acc, false, false);
  }
  return acc;
}

// ---------------------------------------------------------------------------
// WMMA GEMM: C[M x 128] = A[M x 128] @ B[128 x 128], f32 in/out, f16 compute.
// 256 threads = 8 waves; block covers 128 rows; wave covers a 16-row stripe.
// Operands pre-swizzled into fragment layout: each fragment is one aligned
// 32-byte LDS vector load. Stores use a uniform full-tile fast path with all
// addresses folded into immediate offsets.
// ---------------------------------------------------------------------------
__global__ __launch_bounds__(256)
void gemm128_wmma(const float* __restrict__ A, const float* __restrict__ B,
                  float* __restrict__ C, int M) {
  // sAs: A fragments, fragIdx = waveStripe*4 + kt ; 32 frags x 32 lanes x 16 h
  // sBs: B fragments, fragIdx = kt*8 + nt
  __shared__ __attribute__((aligned(32))) _Float16 sAs[32 * 32 * 16];
  __shared__ __attribute__((aligned(32))) _Float16 sBs[32 * 32 * 16];
  const int tid = threadIdx.x;
  const int rowBase = blockIdx.x * 128;

  // Stage 1: coalesced copy of B (row-major, f32 -> f16) into sAs as temp.
  for (int i = tid; i < 128 * 128; i += 256) sAs[i] = (_Float16)B[i];
  __syncthreads();

  // Stage 2: LDS->LDS swizzle of B into fragment layout.
  for (int i = tid; i < 128 * 128; i += 256) {
    int frag = i >> 9;              // kt*8 + nt
    int lane = (i >> 4) & 31;
    int j = i & 15;
    int kt = frag >> 3, nt = frag & 7;
    int k = kt * 32 + kmap(lane >> 4, j);
    int col = nt * 16 + (lane & 15);
    sBs[i] = sAs[k * 128 + col];
  }
  __syncthreads();

  // Stage 3: load A swizzled directly from global into sAs.
  for (int i = tid; i < 128 * 128; i += 256) {
    int frag = i >> 9;              // waveStripe*4 + kt
    int lane = (i >> 4) & 31;
    int j = i & 15;
    int ws = frag >> 2, kt = frag & 3;
    int k = kt * 32 + kmap(lane >> 4, j);
    int row = rowBase + ws * 16 + (lane & 15);
    sAs[i] = (row < M) ? (_Float16)A[(size_t)row * 128 + k] : (_Float16)0.0f;
  }
  __syncthreads();

  const int wave = tid >> 5, lane = tid & 31;
  const int laneRow = lane & 15, kgrp = lane >> 4;
  // First of the 8 output rows this lane owns (D: VGPR r -> row kgrp*8 + r).
  const int rowTop = rowBase + wave * 16 + kgrp * 8;

  v16h afrag[4];
#pragma unroll
  for (int kt = 0; kt < 4; ++kt)
    afrag[kt] = *(const v16h*)&sAs[((wave * 4 + kt) * 32 + lane) * 16];

  float* cp = C + (size_t)rowTop * 128 + laneRow;

  if (rowBase + 128 <= M) {
    // Fast path (all but the last block): unguarded stores, immediate offsets.
#pragma unroll
    for (int nt = 0; nt < 8; ++nt) {
      v8f acc = tile_acc(sBs, afrag, lane, nt);
#pragma unroll
      for (int r = 0; r < 8; ++r) cp[nt * 16 + r * 128] = acc[r];
    }
  } else {
    // Tail block: per-row guard.
#pragma unroll
    for (int nt = 0; nt < 8; ++nt) {
      v8f acc = tile_acc(sBs, afrag, lane, nt);
#pragma unroll
      for (int r = 0; r < 8; ++r)
        if (rowTop + r < M) cp[nt * 16 + r * 128] = acc[r];
    }
  }
}

// ---------------------------------------------------------------------------
// Attention helper kernels
// ---------------------------------------------------------------------------
__device__ __forceinline__ unsigned f2ord(float f) {
  unsigned u = __float_as_uint(f);
  return (u & 0x80000000u) ? ~u : (u | 0x80000000u);
}
__device__ __forceinline__ float ord2f(unsigned u) {
  return (u & 0x80000000u) ? __uint_as_float(u & 0x7fffffffu)
                           : __uint_as_float(~u);
}

__global__ void attn_coef(const float* __restrict__ t,
                          const float* __restrict__ att_src,
                          const float* __restrict__ att_dst,
                          float* __restrict__ asrc, float* __restrict__ adst) {
  int idx = blockIdx.x * blockDim.x + threadIdx.x;
  if (idx >= NNODES * NHEADS) return;
  int node = idx >> 2, h = idx & 3;
  const float* row = t + (size_t)node * NHC + h * NCH;
  float s = 0.f, d = 0.f;
#pragma unroll
  for (int c = 0; c < NCH; ++c) {
    float v = row[c];
    s += v * att_src[h * NCH + c];
    d += v * att_dst[h * NCH + c];
  }
  asrc[idx] = s;
  adst[idx] = d;
}

// econst[h] = sum_c We[0, h*32+c] * att_edge[h, c]  (EDGE_DIM == 1)
__global__ void econst_kernel(const float* __restrict__ We,
                              const float* __restrict__ att_edge,
                              float* __restrict__ econ) {
  int h = threadIdx.x;
  if (h < NHEADS) {
    float s = 0.f;
    for (int c = 0; c < NCH; ++c) s += We[h * NCH + c] * att_edge[h * NCH + c];
    econ[h] = s;
  }
}

__global__ void edge_logits(const int* __restrict__ ei,
                            const float* __restrict__ eattr,
                            const float* __restrict__ asrc,
                            const float* __restrict__ adst,
                            const float* __restrict__ econ,
                            float* __restrict__ buf,
                            unsigned* __restrict__ mbuf) {
  int idx = blockIdx.x * blockDim.x + threadIdx.x;
  if (idx >= NEDGES * NHEADS) return;
  int e = idx >> 2, h = idx & 3;
  int s = ei[e], d = ei[NEDGES + e];
  float l = asrc[s * NHEADS + h] + adst[d * NHEADS + h] + eattr[e] * econ[h];
  l = (l > 0.f) ? l : NEG_SLOPE * l;
  buf[idx] = l;
  atomicMax(&mbuf[d * NHEADS + h], f2ord(l));
}

__global__ void edge_exp(const int* __restrict__ ei, float* __restrict__ buf,
                         const unsigned* __restrict__ mbuf,
                         float* __restrict__ denom) {
  int idx = blockIdx.x * blockDim.x + threadIdx.x;
  if (idx >= NEDGES * NHEADS) return;
  int e = idx >> 2, h = idx & 3;
  int d = ei[NEDGES + e];
  float m = ord2f(mbuf[d * NHEADS + h]);
  float ex = expf(buf[idx] - m);
  buf[idx] = ex;
  atomicAdd(&denom[d * NHEADS + h], ex);
}

__global__ void edge_norm(const int* __restrict__ ei, float* __restrict__ buf,
                          const float* __restrict__ denom,
                          float* __restrict__ alpha_out) {
  int idx = blockIdx.x * blockDim.x + threadIdx.x;
  if (idx >= NEDGES * NHEADS) return;
  int e = idx >> 2, h = idx & 3;
  int d = ei[NEDGES + e];
  float a = buf[idx] / (denom[d * NHEADS + h] + 1e-16f);
  buf[idx] = a;
  if (alpha_out) alpha_out[idx] = a;
}

__global__ void init_out(float* __restrict__ o, const float* __restrict__ bias) {
  int idx = blockIdx.x * blockDim.x + threadIdx.x;
  if (idx < NNODES * NHC) o[idx] = bias[idx & 127];
}

// out[dst] += alpha * h[src] ; thread handles 4 contiguous channels (float4)
__global__ __launch_bounds__(256)
void scatter_msg(const int* __restrict__ ei, const float* __restrict__ alpha,
                 const float* __restrict__ t, float* __restrict__ o) {
  int idx = blockIdx.x * blockDim.x + threadIdx.x;
  if (idx >= NEDGES * 32) return;
  int e = idx >> 5, q = idx & 31;      // q selects a 4-channel chunk, h = q>>3
  int h = q >> 3;
  int s = ei[e], d = ei[NEDGES + e];
  float a = alpha[e * NHEADS + h];
  const float4 tv = *(const float4*)(t + (size_t)s * NHC + q * 4);
  float* op = o + (size_t)d * NHC + q * 4;
  atomicAdd(op + 0, a * tv.x);
  atomicAdd(op + 1, a * tv.y);
  atomicAdd(op + 2, a * tv.z);
  atomicAdd(op + 3, a * tv.w);
}

__global__ void elu_inplace(float* __restrict__ x, int n) {
  int idx = blockIdx.x * blockDim.x + threadIdx.x;
  if (idx < n) {
    float v = x[idx];
    x[idx] = (v > 0.f) ? v : expm1f(v);
  }
}

// ---------------------------------------------------------------------------
// Pooling + MLP head
// ---------------------------------------------------------------------------
__global__ void pool_add(const int* __restrict__ batch,
                         const float* __restrict__ x2,
                         float* __restrict__ pool, float* __restrict__ cnt) {
  int idx = blockIdx.x * blockDim.x + threadIdx.x;
  if (idx >= NNODES * NHC) return;
  int n = idx >> 7, c = idx & 127;
  int g = batch[n];
  atomicAdd(&pool[g * NHC + c], x2[idx]);
  if (c == 0) atomicAdd(&cnt[g], 1.0f);
}

__global__ void mlp1_kernel(const float* __restrict__ pool,
                            const float* __restrict__ cnt,
                            const float* __restrict__ W1,
                            const float* __restrict__ b1,
                            const float* __restrict__ gam,
                            const float* __restrict__ bet,
                            const float* __restrict__ mean,
                            const float* __restrict__ var,
                            float* __restrict__ hid) {
  int idx = blockIdx.x * blockDim.x + threadIdx.x;
  if (idx >= NGRAPH * 64) return;
  int g = idx >> 6, j = idx & 63;
  float inv = 1.0f / fmaxf(cnt[g], 1.0f);
  float s = b1[j];
  for (int c = 0; c < NHC; ++c) s += (pool[g * NHC + c] * inv) * W1[c * 64 + j];
  s = (s - mean[j]) * rsqrtf(var[j] + 1e-5f) * gam[j] + bet[j];
  hid[idx] = fmaxf(s, 0.f);
}

__global__ void mlp2_kernel(const float* __restrict__ hid,
                            const float* __restrict__ W2,
                            const float* __restrict__ b2,
                            float* __restrict__ o) {
  int idx = blockIdx.x * blockDim.x + threadIdx.x;
  if (idx >= NGRAPH * NCLS) return;
  int g = idx >> 2, k = idx & 3;
  float s = b2[k];
  for (int j = 0; j < 64; ++j) s += hid[g * 64 + j] * W2[j * NCLS + k];
  o[idx] = s;
}

__global__ void ei_to_float(const int* __restrict__ ei, float* __restrict__ o) {
  int idx = blockIdx.x * blockDim.x + threadIdx.x;
  if (idx < 2 * NEDGES) o[idx] = (float)ei[idx];
}

// ---------------------------------------------------------------------------
// Host-side layer driver
// ---------------------------------------------------------------------------
static void gat_layer(const float* xin, const float* W, const float* att_s,
                      const float* att_d, const float* We, const float* att_e,
                      const float* bias, const int* eidx, const float* eattr,
                      float* t, float* outbuf, float* asrc, float* adst,
                      unsigned* mbuf, float* denom, float* ebuf, float* econ,
                      float* alpha_out, hipStream_t stream) {
  gemm128_wmma<<<(NNODES + 127) / 128, 256, 0, stream>>>(xin, W, t, NNODES);
  attn_coef<<<(NNODES * NHEADS + 255) / 256, 256, 0, stream>>>(t, att_s, att_d,
                                                               asrc, adst);
  econst_kernel<<<1, 64, 0, stream>>>(We, att_e, econ);
  hipMemsetAsync(mbuf, 0, sizeof(unsigned) * NNODES * NHEADS, stream);
  hipMemsetAsync(denom, 0, sizeof(float) * NNODES * NHEADS, stream);
  edge_logits<<<(NEDGES * NHEADS + 255) / 256, 256, 0, stream>>>(
      eidx, eattr, asrc, adst, econ, ebuf, mbuf);
  edge_exp<<<(NEDGES * NHEADS + 255) / 256, 256, 0, stream>>>(eidx, ebuf, mbuf,
                                                              denom);
  init_out<<<(NNODES * NHC + 255) / 256, 256, 0, stream>>>(outbuf, bias);
  edge_norm<<<(NEDGES * NHEADS + 255) / 256, 256, 0, stream>>>(eidx, ebuf,
                                                               denom, alpha_out);
  scatter_msg<<<(NEDGES * 32 + 255) / 256, 256, 0, stream>>>(eidx, ebuf, t,
                                                             outbuf);
  elu_inplace<<<(NNODES * NHC + 255) / 256, 256, 0, stream>>>(outbuf,
                                                              NNODES * NHC);
}

extern "C" void kernel_launch(void* const* d_in, const int* in_sizes, int n_in,
                              void* d_out, int out_size, void* d_ws,
                              size_t ws_size, hipStream_t stream) {
  (void)in_sizes; (void)n_in; (void)out_size; (void)ws_size;
  const float* x         = (const float*)d_in[0];
  const float* eattr     = (const float*)d_in[1];
  const float* W1        = (const float*)d_in[2];
  const float* att_src1  = (const float*)d_in[3];
  const float* att_dst1  = (const float*)d_in[4];
  const float* We1       = (const float*)d_in[5];
  const float* att_edge1 = (const float*)d_in[6];
  const float* b1        = (const float*)d_in[7];
  const float* W2        = (const float*)d_in[8];
  const float* att_src2  = (const float*)d_in[9];
  const float* att_dst2  = (const float*)d_in[10];
  const float* We2       = (const float*)d_in[11];
  const float* att_edge2 = (const float*)d_in[12];
  const float* b2        = (const float*)d_in[13];
  const float* mlp_W1    = (const float*)d_in[14];
  const float* mlp_b1    = (const float*)d_in[15];
  const float* bn_gamma  = (const float*)d_in[16];
  const float* bn_beta   = (const float*)d_in[17];
  const float* bn_mean   = (const float*)d_in[18];
  const float* bn_var    = (const float*)d_in[19];
  const float* mlp_W2    = (const float*)d_in[20];
  const float* mlp_b2    = (const float*)d_in[21];
  const int*   eidx      = (const int*)d_in[22];
  const int*   batch     = (const int*)d_in[23];
  float* out = (float*)d_out;

  // Workspace layout (floats)
  float* ws = (float*)d_ws;
  size_t o = 0;
  float* t1    = ws + o; o += (size_t)NNODES * NHC;
  float* out1  = ws + o; o += (size_t)NNODES * NHC;   // becomes x1 after elu
  float* t2    = ws + o; o += (size_t)NNODES * NHC;
  float* out2  = ws + o; o += (size_t)NNODES * NHC;   // becomes x2 after elu
  float* asrc  = ws + o; o += (size_t)NNODES * NHEADS;
  float* adst  = ws + o; o += (size_t)NNODES * NHEADS;
  unsigned* mbuf = (unsigned*)(ws + o); o += (size_t)NNODES * NHEADS;
  float* denom = ws + o; o += (size_t)NNODES * NHEADS;
  float* ebuf  = ws + o; o += (size_t)NEDGES * NHEADS; // logits -> exp -> alpha
  float* econ  = ws + o; o += 4;
  float* pool  = ws + o; o += (size_t)NGRAPH * NHC;
  float* cnt   = ws + o; o += (size_t)NGRAPH;
  float* hid   = ws + o; o += (size_t)NGRAPH * 64;

  // Output layout: logits[256] | edge_index as float[2E] | alpha1[E*H]
  float* out_logits = out;
  float* out_ei     = out + NGRAPH * NCLS;
  float* out_alpha  = out_ei + 2 * NEDGES;

  // Layer 1 (alpha written to d_out as part of the reference tuple)
  gat_layer(x, W1, att_src1, att_dst1, We1, att_edge1, b1, eidx, eattr,
            t1, out1, asrc, adst, mbuf, denom, ebuf, econ, out_alpha, stream);
  // Layer 2
  gat_layer(out1, W2, att_src2, att_dst2, We2, att_edge2, b2, eidx, eattr,
            t2, out2, asrc, adst, mbuf, denom, ebuf, econ, nullptr, stream);

  // Global mean pool + MLP head
  hipMemsetAsync(pool, 0, sizeof(float) * NGRAPH * NHC, stream);
  hipMemsetAsync(cnt, 0, sizeof(float) * NGRAPH, stream);
  pool_add<<<(NNODES * NHC + 255) / 256, 256, 0, stream>>>(batch, out2, pool,
                                                           cnt);
  mlp1_kernel<<<(NGRAPH * 64 + 255) / 256, 256, 0, stream>>>(
      pool, cnt, mlp_W1, mlp_b1, bn_gamma, bn_beta, bn_mean, bn_var, hid);
  mlp2_kernel<<<1, 256, 0, stream>>>(hid, mlp_W2, mlp_b2, out_logits);

  // Pass-through edge_index output
  ei_to_float<<<(2 * NEDGES + 255) / 256, 256, 0, stream>>>(eidx, out_ei);
}
